// MultiHeadAttention_31250182046129
// MI455X (gfx1250) — matrix operand order repaired and checked by reading, compiled, not run on hardware
//
#include <hip/hip_runtime.h>
#include <hip/hip_bf16.h>

#define E_DIM  2048
#define H_HEADS 16
#define D_HEAD  128
#define B_SZ    4
#define S_LEN   2048
#define M_ROWS  (B_SZ * S_LEN)   // 8192

typedef _Float16 half_t;
typedef __attribute__((ext_vector_type(16))) _Float16 v16h;
typedef __attribute__((ext_vector_type(8)))  _Float16 v8h;
typedef __attribute__((ext_vector_type(8)))  float    v8f;

// ---------------------------------------------------------------------------
// Fragment loaders matching the CDNA5 16-bit WMMA VGPR layouts (wave32).
//
// A-matrix 16x32 (MxK): lane L<16 holds row M=L, K=0..7 in v0..3, K=16..23 in
// v4..7; lane L>=16 holds row M=L-16, K=8..15 and K=24..31.  => two 16-byte
// contiguous loads per lane from a row-major tile.
static __device__ __forceinline__ v16h load_fragA(const half_t* base, int ld, int lane) {
    const int r  = lane & 15;
    const int hi = (lane >> 4) * 8;
    union { v16h v; v8h h[2]; } u;
    const half_t* p = base + (size_t)r * ld + hi;
    u.h[0] = *(const v8h*)(p);
    u.h[1] = *(const v8h*)(p + 16);
    return u.v;
}

// B-matrix 32x16 (KxN): lane L<16 holds column N=L, K=0..7 and K=16..23;
// lane L>=16 holds column N=L-16, K=8..15 and K=24..31.  For NT GEMM
// (B[k][n] = W[n][k]) that is two 16-byte loads from W row n.
static __device__ __forceinline__ v16h load_fragB(const half_t* base, int ld, int lane) {
    const int n  = lane & 15;
    const int kh = (lane >> 4) * 8;
    union { v16h v; v8h h[2]; } u;
    const half_t* p = base + (size_t)n * ld + kh;
    u.h[0] = *(const v8h*)(p);
    u.h[1] = *(const v8h*)(p + 16);
    return u.v;
}

static __device__ __forceinline__ v8f wmma_f16(v16h a, v16h b, v8f c) {
    return __builtin_amdgcn_wmma_f32_16x16x32_f16(false, a, false, b, (short)0, c, false, false);
}

// ---------------------------------------------------------------------------
// fp32 -> f16 conversion (grid-stride)
__global__ void cvt_f32_f16(const float* __restrict__ in, half_t* __restrict__ out, size_t n) {
    size_t i = (size_t)blockIdx.x * blockDim.x + threadIdx.x;
    size_t stride = (size_t)gridDim.x * blockDim.x;
    for (; i < n; i += stride) out[i] = (half_t)in[i];
}

// ---------------------------------------------------------------------------
// Y = X (M x E, f16) @ W^T (W is ExE row-major, f16) + bias (f32)
// MODE 0: output f16, layout [B][H][S][D]   (for Q and K)
// MODE 1: output f16, layout [B][H][D][S]   (for V, pre-transposed)
// MODE 2: output f32, layout [M][E]         (final output projection)
// Block: 256 threads = 8 waves (4 along M x 2 along N); block tile 128x128,
// wave tile 32x64 = 2x4 WMMA accumulators, K step 32.
template <int MODE>
__global__ void __launch_bounds__(256)
gemm_xwt_bias(const half_t* __restrict__ X, const half_t* __restrict__ W,
              const float* __restrict__ bias, void* __restrict__ outP) {
    const int lane = threadIdx.x & 31;
    const int w    = threadIdx.x >> 5;
    const int wm   = w & 3;
    const int wn   = w >> 2;
    const int rowBase = blockIdx.y * 128 + wm * 32;
    const int colBase = blockIdx.x * 128 + wn * 64;

    v8f acc[2][4];
    for (int i = 0; i < 2; ++i)
        for (int j = 0; j < 4; ++j)
            acc[i][j] = (v8f){};

    for (int kk = 0; kk < E_DIM; kk += 32) {
        v16h a0 = load_fragA(X + (size_t)(rowBase     ) * E_DIM + kk, E_DIM, lane);
        v16h a1 = load_fragA(X + (size_t)(rowBase + 16) * E_DIM + kk, E_DIM, lane);
        v16h b0 = load_fragB(W + (size_t)(colBase     ) * E_DIM + kk, E_DIM, lane);
        v16h b1 = load_fragB(W + (size_t)(colBase + 16) * E_DIM + kk, E_DIM, lane);
        v16h b2 = load_fragB(W + (size_t)(colBase + 32) * E_DIM + kk, E_DIM, lane);
        v16h b3 = load_fragB(W + (size_t)(colBase + 48) * E_DIM + kk, E_DIM, lane);
        acc[0][0] = wmma_f16(a0, b0, acc[0][0]);
        acc[0][1] = wmma_f16(a0, b1, acc[0][1]);
        acc[0][2] = wmma_f16(a0, b2, acc[0][2]);
        acc[0][3] = wmma_f16(a0, b3, acc[0][3]);
        acc[1][0] = wmma_f16(a1, b0, acc[1][0]);
        acc[1][1] = wmma_f16(a1, b1, acc[1][1]);
        acc[1][2] = wmma_f16(a1, b2, acc[1][2]);
        acc[1][3] = wmma_f16(a1, b3, acc[1][3]);
    }

    // C/D layout: lane<16 -> rows v+0..7, lane>=16 -> rows v+8..15; col = lane&15
    const int lhi = lane >> 4;
    const int llo = lane & 15;
    for (int mi = 0; mi < 2; ++mi) {
        for (int ni = 0; ni < 4; ++ni) {
            for (int v = 0; v < 8; ++v) {
                const int r = rowBase + mi * 16 + v + 8 * lhi;
                const int c = colBase + ni * 16 + llo;
                const float val = acc[mi][ni][v] + bias[c];
                if (MODE == 2) {
                    ((float*)outP)[(size_t)r * E_DIM + c] = val;
                } else {
                    const int b = r >> 11;          // / S_LEN
                    const int s = r & (S_LEN - 1);
                    const int h = c >> 7;           // / D_HEAD
                    const int d = c & (D_HEAD - 1);
                    size_t off;
                    if (MODE == 0)
                        off = (((size_t)(b * H_HEADS + h)) * S_LEN + s) * D_HEAD + d;
                    else
                        off = (((size_t)(b * H_HEADS + h)) * D_HEAD + d) * S_LEN + s;
                    ((half_t*)outP)[off] = (half_t)val;
                }
            }
        }
    }
}

// ---------------------------------------------------------------------------
// Causal flash attention.  Q,K: [B][H][S][D] f16.  Vt: [B][H][D][S] f16.
// ctx out: [B][S][E] f16.  Block = 128 threads (4 waves), 64-query tile,
// each wave owns 16 query rows; key tiles of 64; online softmax in f32.
__global__ void __launch_bounds__(128)
flash_attn_causal(const half_t* __restrict__ Q, const half_t* __restrict__ K,
                  const half_t* __restrict__ Vt, half_t* __restrict__ ctx) {
    __shared__ __align__(32) half_t pLds[4 * 16 * 64];   // per-wave 16x64 P tile

    const int lane = threadIdx.x & 31;
    const int wv   = threadIdx.x >> 5;       // wave 0..3
    const int bh   = blockIdx.x;             // b*H + h
    const int b    = bh >> 4;
    const int h    = bh & 15;
    const int qt   = blockIdx.y;             // query tile index
    const int qb   = qt * 64;
    const int qw   = qb + wv * 16;           // first query row of this wave

    const half_t* Qh = Q  + (size_t)bh * S_LEN * D_HEAD;
    const half_t* Kh = K  + (size_t)bh * S_LEN * D_HEAD;
    const half_t* Vh = Vt + (size_t)bh * D_HEAD * S_LEN;

    // Q fragments live in registers for the whole kernel (16 x 128 = 4 k-steps)
    v16h qf[4];
    for (int kd = 0; kd < 4; ++kd)
        qf[kd] = load_fragA(Qh + (size_t)qw * D_HEAD + kd * 32, D_HEAD, lane);

    v8f accO[8];
    for (int i = 0; i < 8; ++i) accO[i] = (v8f){};
    float mrow[8], lrow[8];
    for (int v = 0; v < 8; ++v) { mrow[v] = -3.0e38f; lrow[v] = 0.0f; }

    const int   lhi   = lane >> 4;
    const int   llo   = lane & 15;
    const float scale = 0.08838834764831845f;   // 1/sqrt(128)
    half_t* myLds = pLds + wv * 16 * 64;

    const int ntiles = qt + 1;                  // causal: keys <= qb+63
    for (int kt = 0; kt < ntiles; ++kt) {
        const int kb = kt * 64;

        // S = Q K^T  (rows=queries, cols=keys), 4 col-tiles x 4 k-steps
        v8f sc[4];
        for (int ni = 0; ni < 4; ++ni) sc[ni] = (v8f){};
        for (int ni = 0; ni < 4; ++ni) {
            for (int kd = 0; kd < 4; ++kd) {
                v16h kf = load_fragB(Kh + (size_t)(kb + ni * 16) * D_HEAD + kd * 32,
                                     D_HEAD, lane);
                sc[ni] = wmma_f16(qf[kd], kf, sc[ni]);
            }
        }

        // scale + causal mask + running max
        float nm[8];
        for (int v = 0; v < 8; ++v) {
            const int row = qw + v + 8 * lhi;
            float best = -3.0e38f;
            for (int ni = 0; ni < 4; ++ni) {
                const int col = kb + ni * 16 + llo;
                float sv = sc[ni][v] * scale;
                sv = (col <= row) ? sv : -3.0e38f;
                sc[ni][v] = sv;
                best = fmaxf(best, sv);
            }
            for (int off = 1; off < 16; off <<= 1)
                best = fmaxf(best, __shfl_xor(best, off, 16));
            nm[v] = fmaxf(mrow[v], best);
        }

        // exp, row sums, rescale accumulator
        for (int v = 0; v < 8; ++v) {
            const float cf = __expf(mrow[v] - nm[v]);
            float rs = 0.0f;
            for (int ni = 0; ni < 4; ++ni) {
                const float p = __expf(sc[ni][v] - nm[v]);
                sc[ni][v] = p;
                rs += p;
            }
            for (int off = 1; off < 16; off <<= 1)
                rs += __shfl_xor(rs, off, 16);
            lrow[v] = lrow[v] * cf + rs;
            mrow[v] = nm[v];
            for (int nd = 0; nd < 8; ++nd) accO[nd][v] *= cf;
        }

        // transpose P to A-operand layout through LDS (wave-private tile)
        for (int ni = 0; ni < 4; ++ni)
            for (int v = 0; v < 8; ++v)
                myLds[(v + 8 * lhi) * 64 + ni * 16 + llo] = (half_t)sc[ni][v];
        __syncthreads();

        // O += P V   (P: 16x64 from LDS; V rows contiguous via Vt)
        for (int k2 = 0; k2 < 2; ++k2) {
            v16h pf = load_fragA(myLds + k2 * 32, 64, lane);
            for (int nd = 0; nd < 8; ++nd) {
                v16h vf = load_fragB(Vh + (size_t)(nd * 16) * S_LEN + kb + k2 * 32,
                                     S_LEN, lane);
                accO[nd] = wmma_f16(pf, vf, accO[nd]);
            }
        }
        __syncthreads();
    }

    // normalize and store ctx[b][s][h*D + d] as f16
    for (int v = 0; v < 8; ++v) {
        const float inv  = 1.0f / lrow[v];
        const int   srow = qw + v + 8 * lhi;
        const size_t base = ((size_t)b * S_LEN + srow) * E_DIM + h * D_HEAD;
        for (int nd = 0; nd < 8; ++nd)
            ctx[base + nd * 16 + llo] = (half_t)(accO[nd][v] * inv);
    }
}

// ---------------------------------------------------------------------------
extern "C" void kernel_launch(void* const* d_in, const int* in_sizes, int n_in,
                              void* d_out, int out_size, void* d_ws, size_t ws_size,
                              hipStream_t stream) {
    const float* x  = (const float*)d_in[0];
    const float* Wq = (const float*)d_in[1];
    const float* bq = (const float*)d_in[2];
    const float* Wk = (const float*)d_in[3];
    const float* bk = (const float*)d_in[4];
    const float* Wv = (const float*)d_in[5];
    const float* bv = (const float*)d_in[6];
    const float* Wo = (const float*)d_in[7];
    const float* bo = (const float*)d_in[8];

    char* ws = (char*)d_ws;
    const size_t SZ_X   = (size_t)M_ROWS * E_DIM * sizeof(half_t);  // 32 MiB
    const size_t SZ_W   = (size_t)E_DIM * E_DIM * sizeof(half_t);   //  8 MiB
    half_t* X16  = (half_t*)(ws);
    half_t* W16q = (half_t*)(ws + SZ_X);
    half_t* W16k = (half_t*)(ws + SZ_X + SZ_W);
    half_t* W16v = (half_t*)(ws + SZ_X + 2 * SZ_W);
    half_t* W16o = (half_t*)(ws + SZ_X + 3 * SZ_W);
    half_t* Qb   = (half_t*)(ws + SZ_X + 4 * SZ_W);
    half_t* Kb   = (half_t*)(ws + 2 * SZ_X + 4 * SZ_W);
    half_t* Vtb  = (half_t*)(ws + 3 * SZ_X + 4 * SZ_W);
    half_t* Ctx  = (half_t*)(ws + 4 * SZ_X + 4 * SZ_W);

    // fp32 -> f16 conversions
    cvt_f32_f16<<<2048, 256, 0, stream>>>(x,  X16,  (size_t)M_ROWS * E_DIM);
    cvt_f32_f16<<<1024, 256, 0, stream>>>(Wq, W16q, (size_t)E_DIM * E_DIM);
    cvt_f32_f16<<<1024, 256, 0, stream>>>(Wk, W16k, (size_t)E_DIM * E_DIM);
    cvt_f32_f16<<<1024, 256, 0, stream>>>(Wv, W16v, (size_t)E_DIM * E_DIM);
    cvt_f32_f16<<<1024, 256, 0, stream>>>(Wo, W16o, (size_t)E_DIM * E_DIM);

    // projections
    dim3 ggrid(E_DIM / 128, M_ROWS / 128);   // 16 x 64
    gemm_xwt_bias<0><<<ggrid, 256, 0, stream>>>(X16, W16q, bq, (void*)Qb);
    gemm_xwt_bias<0><<<ggrid, 256, 0, stream>>>(X16, W16k, bk, (void*)Kb);
    gemm_xwt_bias<1><<<ggrid, 256, 0, stream>>>(X16, W16v, bv, (void*)Vtb);

    // causal flash attention
    dim3 fgrid(B_SZ * H_HEADS, S_LEN / 64);  // 64 x 32
    flash_attn_causal<<<fgrid, 128, 0, stream>>>(Qb, Kb, Vtb, Ctx);

    // output projection -> f32 d_out
    gemm_xwt_bias<2><<<ggrid, 256, 0, stream>>>(Ctx, W16o, bo, d_out);
}
